// WorldModel_66305705115653
// MI455X (gfx1250) — matrix operand order repaired and checked by reading, not compiled, not run on hardware
//
#include <hip/hip_runtime.h>
#include <hip/hip_bf16.h>
#include <stdint.h>

// ---------------- model dims ----------------
#define B_    32
#define T_    50
#define A_    6
#define H_    1024
#define E_    1024
#define NC_   32
#define CD_   32
#define SD_   1024           // NC*CD
#define BT_   (B_*T_)        // 1600
#define RNN_K  1030          // SD + A  (true K)
#define RNN_KP 1056          // padded to multiple of 32

typedef __bf16 bf16_t;
typedef __attribute__((ext_vector_type(16))) __bf16 bf16x16;
typedef __attribute__((ext_vector_type(8)))  __bf16 bf16x8;
typedef __attribute__((ext_vector_type(8)))  float  f32x8;

__device__ __forceinline__ bf16_t f2bf(float x) { return (bf16_t)x; }
// branch-free fast activations (native exp/log/div, no libm exec-divergence)
__device__ __forceinline__ float eluf(float x) { return x > 0.f ? x : __expf(x) - 1.f; }
__device__ __forceinline__ float sigm(float x) { return __fdividef(1.f, 1.f + __expf(-x)); }
__device__ __forceinline__ float tanhfast(float x)
{
    float xc = fminf(fmaxf(x, -15.f), 15.f);
    float e  = __expf(-2.f * xc);
    return (1.f - e) * __fdividef(1.f, 1.f + e);
}

__device__ __forceinline__ bf16x16 cat8(bf16x8 a, bf16x8 b)
{
    return __builtin_shufflevector(a, b, 0, 1, 2, 3, 4, 5, 6, 7,
                                         8, 9, 10, 11, 12, 13, 14, 15);
}

// =====================================================================
// Guard-free wave-level WMMA GEMM: C[M,N] = A[M,K] * W[N,K]^T + bias.
// REQUIRES: M % 16 == 0, N % (16*NT) == 0, K % 32 == 0, lda/ldw even.
// One wave computes a 16 x (16*NT) strip: one A-fragment pair feeds NT
// WMMAs per K-step (register blocking along N).
// =====================================================================
template<bool STORE_BF16, int NT>
__global__ void gemm_ws(const bf16_t* __restrict__ A, int lda,
                        const bf16_t* __restrict__ W, int ldw,
                        const float*  __restrict__ bias,
                        void* __restrict__ Cp, int ldc,
                        int M, int N, int K)
{
    const int tilesM = M >> 4;
    const int stripsN = N / (16 * NT);
    const int nWork  = tilesM * stripsN;
    const int wave   = blockIdx.x * (blockDim.x >> 5) + (threadIdx.x >> 5);
    if (wave >= nWork) return;                  // wave-uniform exit
    const int tm   = wave % tilesM;
    const int ts   = wave / tilesM;
    const int lane = threadIdx.x & 31;
    const int hi   = lane >> 4;
    const int lo   = lane & 15;
    const int m    = tm * 16 + lo;

    const bf16_t* Ap = A + (size_t)m * lda + hi * 8;     // K = k0+{0..7,16..23}+8*hi
    const bf16_t* Wp[NT];
#pragma unroll
    for (int u = 0; u < NT; ++u)
        Wp[u] = W + (size_t)(ts * 16 * NT + u * 16 + lo) * ldw + hi * 8;

    f32x8 acc[NT];
#pragma unroll
    for (int u = 0; u < NT; ++u) acc[u] = (f32x8){};

    for (int k0 = 0; k0 < K; k0 += 32) {
        __builtin_prefetch(Wp[0] + k0 + 256, 0, 1);      // global_prefetch_b8
        bf16x8 a0 = *(const bf16x8*)(Ap + k0);
        bf16x8 a1 = *(const bf16x8*)(Ap + k0 + 16);
        bf16x16 af = cat8(a0, a1);
#pragma unroll
        for (int u = 0; u < NT; ++u) {
            bf16x8 b0 = *(const bf16x8*)(Wp[u] + k0);
            bf16x8 b1 = *(const bf16x8*)(Wp[u] + k0 + 16);
            acc[u] = __builtin_amdgcn_wmma_f32_16x16x32_bf16(false, af,
                                                             false, cat8(b0, b1),
                                                             (short)0, acc[u], false, false);
        }
    }
#pragma unroll
    for (int u = 0; u < NT; ++u) {
        const int nn = ts * 16 * NT + u * 16 + lo;
        const float bv = bias[nn];
#pragma unroll
        for (int r = 0; r < 8; ++r) {
            int mm = tm * 16 + r + hi * 8;   // C layout: VGPR r, hi half = rows +8
            float v = acc[u][r] + bv;
            if (STORE_BF16) ((bf16_t*)Cp)[(size_t)mm * ldc + nn] = f2bf(v);
            else            ((float*) Cp)[(size_t)mm * ldc + nn] = v;
        }
    }
}

// =====================================================================
// Implicit-GEMM conv: 4x4 kernel, stride 2, pad 1, NCHW, fused bias+ELU,
// bf16 output. One wave computes MT output-channel tiles for one 16-pixel
// tile: the expensive im2col B-gather is reused across MT WMMAs per K-step.
// Halo handled by clamped load + select (branch-free).
// =====================================================================
template<typename Tin, int CIN, int HIN, int WIN, int COUT, int KP, int MT>
__global__ void conv_gemm(const Tin* __restrict__ in, const bf16_t* __restrict__ w,
                          const float* __restrict__ bias, bf16_t* __restrict__ out)
{
    constexpr int HOUT = HIN / 2, WOUT = WIN / 2;
    constexpr int K    = CIN * 16;
    constexpr int NTOT = BT_ * HOUT * WOUT;      // multiple of 16 for all layers
    constexpr int groupsM = COUT / (16 * MT);
    constexpr int tilesN  = NTOT / 16;
    const int wave = blockIdx.x * (blockDim.x >> 5) + (threadIdx.x >> 5);
    if (wave >= groupsM * tilesN) return;
    const int gm = wave % groupsM, tn = wave / groupsM;
    const int lane = threadIdx.x & 31, hi = lane >> 4, lo = lane & 15;
    const int n = tn * 16 + lo;                  // output pixel
    const int img = n / (HOUT * WOUT);
    const int rem = n % (HOUT * WOUT);
    const int oy  = rem / WOUT, ox = rem % WOUT;
    const Tin* inImg = in + (size_t)img * CIN * HIN * WIN;

    const bf16_t* Ap[MT];
#pragma unroll
    for (int u = 0; u < MT; ++u)
        Ap[u] = w + (size_t)((gm * MT + u) * 16 + lo) * KP + hi * 8;

    f32x8 acc[MT];
#pragma unroll
    for (int u = 0; u < MT; ++u) acc[u] = (f32x8){};

    for (int k0 = 0; k0 < KP; k0 += 32) {
        bf16x16 bq;
#pragma unroll
        for (int e = 0; e < 16; ++e) {
            int kb = k0 + hi * 8 + (e < 8 ? e : e + 8);
            int ci = kb >> 4, rr = kb & 15, ky = rr >> 2, kx = rr & 3;
            int iy = oy * 2 - 1 + ky, ix = ox * 2 - 1 + kx;
            bool inb = ((unsigned)iy < (unsigned)HIN) &&
                       ((unsigned)ix < (unsigned)WIN) && (kb < K);
            // clamped address, unconditional load, select (no exec branches)
            int cic = inb ? ci : 0, iyc = inb ? iy : 0, ixc = inb ? ix : 0;
            float v = (float)inImg[(size_t)cic * (HIN * WIN) + iyc * WIN + ixc];
            bq[e] = f2bf(inb ? v : 0.f);
        }
#pragma unroll
        for (int u = 0; u < MT; ++u) {
            bf16x8 a0 = *(const bf16x8*)(Ap[u] + k0);
            bf16x8 a1 = *(const bf16x8*)(Ap[u] + k0 + 16);
            acc[u] = __builtin_amdgcn_wmma_f32_16x16x32_bf16(false, cat8(a0, a1),
                                                             false, bq,
                                                             (short)0, acc[u], false, false);
        }
    }
#pragma unroll
    for (int u = 0; u < MT; ++u) {
#pragma unroll
        for (int r = 0; r < 8; ++r) {
            int mm = (gm * MT + u) * 16 + r + hi * 8;
            float v = eluf(acc[u][r] + bias[mm]);
            out[((size_t)img * COUT + mm) * (HOUT * WOUT) + rem] = f2bf(v);
        }
    }
}

// ---------------- elementwise helpers ----------------
// f32 -> bf16 with row padding: dst rows of Kp, zero for k >= K.
__global__ void cvt_pad(const float* __restrict__ s, bf16_t* __restrict__ d,
                        int rows, int K, int Kp)
{
    int i = blockIdx.x * blockDim.x + threadIdx.x;
    if (i >= rows * Kp) return;
    int r = i / Kp, k = i - r * Kp;
    d[i] = f2bf(k < K ? s[(size_t)r * K + k] : 0.f);
}

__global__ void init_state(const float* __restrict__ prevH, const float* __restrict__ prevS,
                           float* __restrict__ hidden, bf16_t* __restrict__ h_bf,
                           bf16_t* __restrict__ rnn_in)
{
    int i = blockIdx.x * blockDim.x + threadIdx.x;     // B_*RNN_KP
    if (i >= B_ * RNN_KP) return;
    int b = i / RNN_KP, j = i - b * RNN_KP;
    if (j < SD_) {
        rnn_in[i] = f2bf(prevS[(size_t)b * SD_ + j]);
        float h = prevH[(size_t)b * H_ + j];
        hidden[(size_t)b * H_ + j] = h;
        h_bf[(size_t)b * H_ + j]   = f2bf(h);
    } else {
        rnn_in[i] = f2bf(0.f);                          // action + pad region
    }
}

__global__ void prep_step(const float* __restrict__ actions, const bf16_t* __restrict__ emb_b,
                          bf16_t* __restrict__ rnn_in, bf16_t* __restrict__ post_in, int t)
{
    int i = blockIdx.x * blockDim.x + threadIdx.x;
    if (i >= B_ * E_) return;
    int b = i >> 10, j = i & 1023;
    post_in[(size_t)b * (H_ + E_) + H_ + j] = emb_b[((size_t)b * T_ + t) * E_ + j];
    if (j < A_)
        rnn_in[(size_t)b * RNN_KP + SD_ + j] = f2bf(actions[((size_t)b * T_ + t) * A_ + j]);
}

__global__ void gru_pointwise(const float* __restrict__ gi, const float* __restrict__ gh,
                              float* __restrict__ hidden, bf16_t* __restrict__ h_bf,
                              bf16_t* __restrict__ post_in, float* __restrict__ hs_out, int t)
{
    int i = blockIdx.x * blockDim.x + threadIdx.x;
    if (i >= B_ * H_) return;
    int b = i >> 10, j = i & 1023;
    const float* gir = gi + (size_t)b * 3 * H_;
    const float* ghr = gh + (size_t)b * 3 * H_;
    float r  = sigm(gir[j] + ghr[j]);
    float z  = sigm(gir[H_ + j] + ghr[H_ + j]);
    float nn = tanhfast(gir[2 * H_ + j] + r * ghr[2 * H_ + j]);
    float h  = (1.f - z) * nn + z * hidden[i];
    hidden[i] = h;
    h_bf[i]   = f2bf(h);
    post_in[(size_t)b * (H_ + E_) + j] = f2bf(h);
    hs_out[((size_t)b * T_ + t) * H_ + j] = h;
}

__global__ void ln_elu(const float* __restrict__ x, const float* __restrict__ g,
                       const float* __restrict__ be, bf16_t* __restrict__ y)
{
    __shared__ float s1[256], s2[256];
    const int b = blockIdx.x, tid = threadIdx.x;
    const float* xr = x + (size_t)b * H_;
    float a0 = xr[tid], a1 = xr[tid + 256], a2 = xr[tid + 512], a3 = xr[tid + 768];
    s1[tid] = a0 + a1 + a2 + a3;
    s2[tid] = a0 * a0 + a1 * a1 + a2 * a2 + a3 * a3;
    __syncthreads();
    for (int o = 128; o > 0; o >>= 1) {
        if (tid < o) { s1[tid] += s1[tid + o]; s2[tid] += s2[tid + o]; }
        __syncthreads();
    }
    float mean = s1[0] * (1.f / H_);
    float var  = s2[0] * (1.f / H_) - mean * mean;
    float rstd = rsqrtf(var + 1e-5f);
#pragma unroll
    for (int u = 0; u < 4; ++u) {
        int j = tid + u * 256;
        float v = (xr[j] - mean) * rstd * g[j] + be[j];
        y[(size_t)b * H_ + j] = f2bf(eluf(v));
    }
}

__device__ __forceinline__ uint32_t hashu(uint32_t x)
{
    x ^= x >> 16; x *= 0x7feb352du;
    x ^= x >> 15; x *= 0x846ca68bu;
    x ^= x >> 16; return x;
}

// Gumbel-argmax categorical sample -> one-hot, written to ps output and as
// next-step recurrent state input (forward value of straight-through sample).
__global__ void sample_onehot(const float* __restrict__ ql_base, float* __restrict__ ps_base,
                              bf16_t* __restrict__ rnn_in, int t)
{
    int i = blockIdx.x * blockDim.x + threadIdx.x;   // B*NC = 1024
    if (i >= B_ * NC_) return;
    int b = i >> 5, c = i & 31;
    const float* lg = ql_base + ((size_t)b * T_ + t) * (NC_ * CD_) + c * CD_;
    int best = 0; float bestv = -3.4e38f;
    for (int j = 0; j < CD_; ++j) {
        uint32_t h = hashu(((uint32_t)t * 0x9E3779B9u) ^ (uint32_t)(i * CD_ + j) ^ 0xC0FFEEu);
        float u = ((float)(h >> 8) + 0.5f) * (1.f / 16777216.f);
        float v = lg[j] - __logf(-__logf(u));
        if (v > bestv) { bestv = v; best = j; }
    }
    float*  po = ps_base + ((size_t)b * T_ + t) * (NC_ * CD_) + c * CD_;
    bf16_t* ri = rnn_in + (size_t)b * RNN_KP + c * CD_;
    for (int j = 0; j < CD_; ++j) {
        float oh = (j == best) ? 1.f : 0.f;
        po[j] = oh;
        ri[j] = f2bf(oh);
    }
}

// =====================================================================
extern "C" void kernel_launch(void* const* d_in, const int* in_sizes, int n_in,
                              void* d_out, int out_size, void* d_ws, size_t ws_size,
                              hipStream_t stream)
{
    (void)in_sizes; (void)n_in; (void)out_size;
    const float* obs  = (const float*)d_in[0];
    const float* acts = (const float*)d_in[1];
    const float* prevH = (const float*)d_in[2];
    const float* prevS = (const float*)d_in[3];
    const float* cw1 = (const float*)d_in[4],  *cb1 = (const float*)d_in[5];
    const float* cw2 = (const float*)d_in[6],  *cb2 = (const float*)d_in[7];
    const float* cw3 = (const float*)d_in[8],  *cb3 = (const float*)d_in[9];
    const float* cw4 = (const float*)d_in[10], *cb4 = (const float*)d_in[11];
    const float* ew  = (const float*)d_in[12], *eb  = (const float*)d_in[13];
    const float* wih = (const float*)d_in[14], *whh = (const float*)d_in[15];
    const float* bih = (const float*)d_in[16], *bhh = (const float*)d_in[17];
    const float* pw1 = (const float*)d_in[18], *pb1 = (const float*)d_in[19];
    const float* pg  = (const float*)d_in[20], *pbe = (const float*)d_in[21];
    const float* pw2 = (const float*)d_in[22], *pb2 = (const float*)d_in[23];
    const float* qw1 = (const float*)d_in[24], *qb1 = (const float*)d_in[25];
    const float* qg  = (const float*)d_in[26], *qbe = (const float*)d_in[27];
    const float* qw2 = (const float*)d_in[28], *qb2 = (const float*)d_in[29];

    const size_t OUT1 = (size_t)B_ * T_ * NC_ * CD_;   // 1,638,400
    float* pl = (float*)d_out;
    float* ql = pl + OUT1;
    float* ps = ql + OUT1;
    float* hs = ps + OUT1;

    // ---- workspace carve ----
    char*  base = (char*)d_ws;
    size_t off  = 0;
    auto carve = [&](size_t elems, size_t esz) -> void* {
        void* p = base + off;
        off += (elems * esz + 255) & ~(size_t)255;
        return p;
    };
    bf16_t* cw1b = (bf16_t*)carve(32 * 64, 2);                 // padded K 48->64
    bf16_t* cw2b = (bf16_t*)carve(64 * 512, 2);
    bf16_t* cw3b = (bf16_t*)carve(128 * 1024, 2);
    bf16_t* cw4b = (bf16_t*)carve(256 * 2048, 2);
    bf16_t* ewb  = (bf16_t*)carve((size_t)1024 * 4096, 2);
    bf16_t* wihb = (bf16_t*)carve((size_t)3072 * RNN_KP, 2);   // padded K 1030->1056
    bf16_t* whhb = (bf16_t*)carve((size_t)3072 * 1024, 2);
    bf16_t* pw1b = (bf16_t*)carve((size_t)1024 * 1024, 2);
    bf16_t* pw2b = (bf16_t*)carve((size_t)1024 * 1024, 2);
    bf16_t* qw1b = (bf16_t*)carve((size_t)1024 * 2048, 2);
    bf16_t* qw2b = (bf16_t*)carve((size_t)1024 * 1024, 2);
    bf16_t* act1 = (bf16_t*)carve((size_t)BT_ * 32 * 32 * 32, 2);
    bf16_t* act2 = (bf16_t*)carve((size_t)BT_ * 64 * 16 * 16, 2);
    bf16_t* act3 = (bf16_t*)carve((size_t)BT_ * 128 * 8 * 8, 2);
    bf16_t* act4 = (bf16_t*)carve((size_t)BT_ * 256 * 4 * 4, 2);
    bf16_t* embb = (bf16_t*)carve((size_t)BT_ * E_, 2);
    bf16_t* rnn_in  = (bf16_t*)carve((size_t)B_ * RNN_KP, 2);
    float*  hidden  = (float*) carve((size_t)B_ * H_, 4);
    bf16_t* h_bf    = (bf16_t*)carve((size_t)B_ * H_, 2);
    float*  gi      = (float*) carve((size_t)B_ * 3 * H_, 4);
    float*  gh      = (float*) carve((size_t)B_ * 3 * H_, 4);
    float*  x1      = (float*) carve((size_t)B_ * H_, 4);
    bf16_t* x1b     = (bf16_t*)carve((size_t)B_ * H_, 2);
    bf16_t* post_in = (bf16_t*)carve((size_t)B_ * (H_ + E_), 2);
    float*  q1      = (float*) carve((size_t)B_ * H_, 4);
    bf16_t* q1b     = (bf16_t*)carve((size_t)B_ * H_, 2);
    if (off > ws_size) return;   // workspace too small; nothing safe to do

    auto cvt = [&](const float* s, bf16_t* d, int rows, int K, int Kp) {
        size_t n = (size_t)rows * Kp;
        cvt_pad<<<dim3((unsigned)((n + 255) / 256)), dim3(256), 0, stream>>>(s, d, rows, K, Kp);
    };
    cvt(cw1, cw1b, 32, 48, 64);
    cvt(cw2, cw2b, 64, 512, 512);
    cvt(cw3, cw3b, 128, 1024, 1024);
    cvt(cw4, cw4b, 256, 2048, 2048);
    cvt(ew,  ewb,  1024, 4096, 4096);
    cvt(wih, wihb, 3072, RNN_K, RNN_KP);
    cvt(whh, whhb, 3072, 1024, 1024);
    cvt(pw1, pw1b, 1024, 1024, 1024);
    cvt(pw2, pw2b, 1024, 1024, 1024);
    cvt(qw1, qw1b, 1024, 2048, 2048);
    cvt(qw2, qw2b, 1024, 1024, 1024);

    // ---- conv encoder (implicit-GEMM WMMA, MT channel tiles per wave) ----
    conv_gemm<float, 3, 64, 64, 32, 64, 2>
        <<<(1 * (BT_ * 32 * 32 / 16) + 7) / 8, 256, 0, stream>>>(obs, cw1b, cb1, act1);
    conv_gemm<bf16_t, 32, 32, 32, 64, 512, 4>
        <<<(1 * (BT_ * 16 * 16 / 16) + 7) / 8, 256, 0, stream>>>(act1, cw2b, cb2, act2);
    conv_gemm<bf16_t, 64, 16, 16, 128, 1024, 4>
        <<<(2 * (BT_ * 8 * 8 / 16) + 7) / 8, 256, 0, stream>>>(act2, cw3b, cb3, act3);
    conv_gemm<bf16_t, 128, 8, 8, 256, 2048, 4>
        <<<(4 * (BT_ * 4 * 4 / 16) + 7) / 8, 256, 0, stream>>>(act3, cw4b, cb4, act4);

    auto ggrid = [](int M, int N) { return ((M >> 4) * (N >> 6) + 7) / 8; };  // NT=4
    // embed: [1600,4096] @ ew^T + eb -> bf16 [1600,1024]
    gemm_ws<true, 4><<<ggrid(BT_, E_), 256, 0, stream>>>(
        act4, 4096, ewb, 4096, eb, embb, E_, BT_, E_, 4096);

    // ---- recurrent scan ----
    init_state<<<(B_ * RNN_KP + 255) / 256, 256, 0, stream>>>(prevH, prevS, hidden, h_bf, rnn_in);

    for (int t = 0; t < T_; ++t) {
        prep_step<<<(B_ * E_ + 255) / 256, 256, 0, stream>>>(acts, embb, rnn_in, post_in, t);
        // gi = rnn_in @ wih^T + bih   [32,3072], K padded to 1056
        gemm_ws<false, 4><<<ggrid(B_, 3 * H_), 256, 0, stream>>>(
            rnn_in, RNN_KP, wihb, RNN_KP, bih, gi, 3 * H_, B_, 3 * H_, RNN_KP);
        // gh = hidden @ whh^T + bhh   [32,3072]
        gemm_ws<false, 4><<<ggrid(B_, 3 * H_), 256, 0, stream>>>(
            h_bf, H_, whhb, H_, bhh, gh, 3 * H_, B_, 3 * H_, H_);
        gru_pointwise<<<(B_ * H_ + 255) / 256, 256, 0, stream>>>(
            gi, gh, hidden, h_bf, post_in, hs, t);
        // prior: x1 = h @ pw1^T + pb1; LN+ELU; pl_t = x1b @ pw2^T + pb2
        gemm_ws<false, 4><<<ggrid(B_, H_), 256, 0, stream>>>(
            h_bf, H_, pw1b, H_, pb1, x1, H_, B_, H_, H_);
        ln_elu<<<B_, 256, 0, stream>>>(x1, pg, pbe, x1b);
        gemm_ws<false, 4><<<ggrid(B_, H_), 256, 0, stream>>>(
            x1b, H_, pw2b, H_, pb2, pl + (size_t)t * (NC_ * CD_), T_ * NC_ * CD_,
            B_, NC_ * CD_, H_);
        // posterior: q1 = [h, emb] @ qw1^T + qb1; LN+ELU; ql_t = q1b @ qw2^T + qb2
        gemm_ws<false, 4><<<ggrid(B_, H_), 256, 0, stream>>>(
            post_in, H_ + E_, qw1b, H_ + E_, qb1, q1, H_, B_, H_, H_ + E_);
        ln_elu<<<B_, 256, 0, stream>>>(q1, qg, qbe, q1b);
        gemm_ws<false, 4><<<ggrid(B_, H_), 256, 0, stream>>>(
            q1b, H_, qw2b, H_, qb2, ql + (size_t)t * (NC_ * CD_), T_ * NC_ * CD_,
            B_, NC_ * CD_, H_);
        // sample one-hot -> ps output and next-step recurrent state input
        sample_onehot<<<(B_ * NC_ + 255) / 256, 256, 0, stream>>>(ql, ps, rnn_in, t);
    }
}
